// GNT_76235669504564
// MI455X (gfx1250) — matrix operand order repaired and checked by reading, not compiled, not run on hardware
//
#include <hip/hip_runtime.h>
#include <math.h>

// ---------------------------------------------------------------------------
// GNT forward for MI455X (gfx1250). All linears go through WMMA
// (v_wmma_f32_16x16x32_f16, f32 accumulate), wave32.
// Weights pre-packed into the ISA 7.12.2 B-fragment layout (f16, zero padded)
// so GEMM inner loops are branch-free with clause'd b128 loads. The wide
// kernel computes a 16x64 strip per wave: one A fragment feeds 4 WMMAs.
// ---------------------------------------------------------------------------

#define R_     256
#define S_     64
#define V_     10
#define DIM_   64
#define FF_    256
#define NH_    4
#define SEM_   512
#define DEPTH_ 8
#define NFREQ_ 10
#define PE_    63
#define NRS_   (R_ * S_)        /* 16384  */
#define NRSV_  (R_ * S_ * V_)   /* 163840 */

typedef _Float16 f16;
typedef __attribute__((ext_vector_type(16))) _Float16 v16h;
typedef __attribute__((ext_vector_type(8)))  float    v8f;

__device__ __forceinline__ v8f wmma_f16(v16h a, v16h b, v8f c)
{
    return __builtin_amdgcn_wmma_f32_16x16x32_f16(false, a, false, b,
                                                  (short)0, c, false, false);
}

// ---------------------------------------------------------------------------
// Weight packer: W f32 row-major [K,N] -> f16 B-fragments, zero padded.
// Entry e = (tn*nslab + sl)*32 + lane holds the 16 values lane needs for
// K-slab sl of column tile tn:  b[i] <-> k = sl*32 + (lane>=16?8:0) + i + (i>=8?8:0),
// n = tn*16 + (lane&15).  One thread per entry (writes 32 bytes).
// ---------------------------------------------------------------------------
__global__ void __launch_bounds__(256)
k_packW(const float* __restrict__ W, f16* __restrict__ out,
        int K, int N, int nslab, long nentries)
{
    long t = (long)blockIdx.x * 256 + threadIdx.x;
    if (t >= nentries) return;
    int lane = (int)(t & 31);
    long rest = t >> 5;
    int sl = (int)(rest % nslab);
    int tn = (int)(rest / nslab);
    int half = lane >> 4, lm = lane & 15;
    int n = tn * 16 + lm;
    f16* o = out + t * 16;
#pragma unroll
    for (int i = 0; i < 16; ++i) {
        int k = sl * 32 + half * 8 + i + (i >= 8 ? 8 : 0);
        float v = (k < K && n < N) ? W[(long)k * N + n] : 0.f;
        o[i] = (f16)v;
    }
}

// ---------------------------------------------------------------------------
// Packed-weight GEMM, 1 tile (16x16) per wave. Used only when N % 64 != 0.
// Tail slab: A addresses clamped in-bounds (real finite data), packed B is
// zero there, so extra products contribute exactly 0.
// ---------------------------------------------------------------------------
__global__ void __launch_bounds__(256)
k_gemm_pw(const float* __restrict__ A, long lda,
          const f16* __restrict__ Wp, int nslab,
          const float* __restrict__ bias,
          const float* __restrict__ Res, long ldr,
          float* __restrict__ Cf, f16* __restrict__ Ch, long ldc,
          int M, int N, int K, int relu)
{
    const int lane = threadIdx.x & 31;
    const int half = lane >> 4;
    const int lm   = lane & 15;
    const int tM = (M + 15) >> 4;
    const int tN = (N + 15) >> 4;
    long tile = (long)blockIdx.x * 8 + (threadIdx.x >> 5);
    if (tile >= (long)tM * tN) return;               // wave-uniform exit
    const int tm = (int)(tile / tN);
    const int tn = (int)(tile % tN);

    int row = tm * 16 + lm; if (row >= M) row = M - 1;
    const float* Arow = A + (long)row * lda;
    const v16h* Bp = ((const v16h*)Wp) + ((long)tn * nslab) * 32 + lane;

    v8f acc = {0.f, 0.f, 0.f, 0.f, 0.f, 0.f, 0.f, 0.f};
    const int Kmain = K & ~31;
    int sl = 0;
    for (int k0 = 0; k0 < Kmain; k0 += 32, ++sl) {
        const float* a0 = Arow + k0 + half * 8;
        v16h av;
#pragma unroll
        for (int i = 0; i < 8; ++i) {
            av[i]     = (f16)a0[i];
            av[i + 8] = (f16)a0[16 + i];
        }
        v16h bv = Bp[(long)sl * 32];
        acc = wmma_f16(av, bv, acc);
    }
    if (Kmain < K) {                                  // single padded tail slab
        v16h av;
#pragma unroll
        for (int i = 0; i < 8; ++i) {
            int ka = Kmain + half * 8 + i;      if (ka >= K) ka = K - 1;
            int kb = Kmain + 16 + half * 8 + i; if (kb >= K) kb = K - 1;
            av[i]     = (f16)Arow[ka];
            av[i + 8] = (f16)Arow[kb];
        }
        v16h bv = Bp[(long)sl * 32];                  // zero-padded past K
        acc = wmma_f16(av, bv, acc);
    }

    const int n = tn * 16 + lm;
    if (n >= N) return;
    const float bn = bias ? bias[n] : 0.f;
#pragma unroll
    for (int j = 0; j < 8; ++j) {
        int m = tm * 16 + half * 8 + j;
        if (m >= M) continue;
        float v = acc[j] + bn;
        if (Res)  v += Res[(long)m * ldr + n];
        if (relu) v = fmaxf(v, 0.f);
        if (Cf) Cf[(long)m * ldc + n] = v;
        else    Ch[(long)m * ldc + n] = (f16)v;
    }
}

// ---------------------------------------------------------------------------
// Packed-weight GEMM, 16x64 strip per wave (requires N % 64 == 0).
// One A fragment per K-slab feeds 4 independent WMMAs (no D->A/B hazards),
// amortizing A loads and f32->f16 conversion 4x.
// ---------------------------------------------------------------------------
__global__ void __launch_bounds__(256)
k_gemm_pw4(const float* __restrict__ A, long lda,
           const f16* __restrict__ Wp, int nslab,
           const float* __restrict__ bias,
           const float* __restrict__ Res, long ldr,
           float* __restrict__ Cf, f16* __restrict__ Ch, long ldc,
           int M, int N, int K, int relu)
{
    const int lane = threadIdx.x & 31;
    const int half = lane >> 4;
    const int lm   = lane & 15;
    const int tM  = (M + 15) >> 4;
    const int tN4 = N >> 6;
    long tg = (long)blockIdx.x * 8 + (threadIdx.x >> 5);
    if (tg >= (long)tM * tN4) return;                // wave-uniform exit
    const int tm  = (int)(tg / tN4);
    const int tn0 = (int)(tg % tN4) * 4;

    int row = tm * 16 + lm; if (row >= M) row = M - 1;
    const float* Arow = A + (long)row * lda;
    const long tstride = (long)nslab * 32;           // v16h entries per tile
    const v16h* Bp = ((const v16h*)Wp) + (long)tn0 * tstride + lane;

    v8f acc[4];
#pragma unroll
    for (int t = 0; t < 4; ++t) acc[t] = v8f{0.f,0.f,0.f,0.f,0.f,0.f,0.f,0.f};

    const int Kmain = K & ~31;
    int sl = 0;
    for (int k0 = 0; k0 < Kmain; k0 += 32, ++sl) {
        const float* a0 = Arow + k0 + half * 8;
        v16h av;
#pragma unroll
        for (int i = 0; i < 8; ++i) {
            av[i]     = (f16)a0[i];
            av[i + 8] = (f16)a0[16 + i];
        }
        const v16h* b = Bp + (long)sl * 32;
#pragma unroll
        for (int t = 0; t < 4; ++t)
            acc[t] = wmma_f16(av, b[t * tstride], acc[t]);
    }
    if (Kmain < K) {                                  // single padded tail slab
        v16h av;
#pragma unroll
        for (int i = 0; i < 8; ++i) {
            int ka = Kmain + half * 8 + i;      if (ka >= K) ka = K - 1;
            int kb = Kmain + 16 + half * 8 + i; if (kb >= K) kb = K - 1;
            av[i]     = (f16)Arow[ka];
            av[i + 8] = (f16)Arow[kb];
        }
        const v16h* b = Bp + (long)sl * 32;
#pragma unroll
        for (int t = 0; t < 4; ++t)
            acc[t] = wmma_f16(av, b[t * tstride], acc[t]);
    }

#pragma unroll
    for (int t = 0; t < 4; ++t) {
        const int n = (tn0 + t) * 16 + lm;
        const float bn = bias ? bias[n] : 0.f;
#pragma unroll
        for (int j = 0; j < 8; ++j) {
            int m = tm * 16 + half * 8 + j;
            if (m >= M) continue;
            float v = acc[t][j] + bn;
            if (Res)  v += Res[(long)m * ldr + n];
            if (relu) v = fmaxf(v, 0.f);
            if (Cf) Cf[(long)m * ldc + n] = v;
            else    Ch[(long)m * ldc + n] = (f16)v;
        }
    }
}

// ---------------------------------------------------------------------------
// Generic batched GEMM (f32 A and W) for the attention products.
// Branch-free: main loop unconditional; tail uses clamped loads + selects.
// W element [k,n] at W[k*ldw+n] (wtrans=0) or W[n*ldw+k] (wtrans=1).
// ---------------------------------------------------------------------------
__global__ void __launch_bounds__(256)
k_gemm(const float* __restrict__ A, long lda, long sAy, long sAz,
       const float* __restrict__ W, long ldw, long sWy, long sWz, int wtrans,
       float* __restrict__ Cf, long ldc, long sCy, long sCz,
       int M, int N, int K, float scale)
{
    const int lane = threadIdx.x & 31;
    const int half = lane >> 4;
    const int lm   = lane & 15;
    const int tM = (M + 15) >> 4;
    const int tN = (N + 15) >> 4;
    long tile = (long)blockIdx.x * 8 + (threadIdx.x >> 5);
    if (tile >= (long)tM * tN) return;
    const int tm = (int)(tile / tN);
    const int tn = (int)(tile % tN);

    const float* Ab = A + (long)blockIdx.y * sAy + (long)blockIdx.z * sAz;
    const float* Wb = W + (long)blockIdx.y * sWy + (long)blockIdx.z * sWz;

    int row = tm * 16 + lm; if (row >= M) row = M - 1;
    int col = tn * 16 + lm; if (col >= N) col = N - 1;
    const float* Arow = Ab + (long)row * lda;

    v8f acc = {0.f, 0.f, 0.f, 0.f, 0.f, 0.f, 0.f, 0.f};
    const int Kmain = K & ~31;
    for (int k0 = 0; k0 < Kmain; k0 += 32) {
        v16h av, bv;
#pragma unroll
        for (int i = 0; i < 8; ++i) {
            av[i]     = (f16)Arow[k0 + half * 8 + i];
            av[i + 8] = (f16)Arow[k0 + 16 + half * 8 + i];
        }
#pragma unroll
        for (int i = 0; i < 16; ++i) {
            int kw = k0 + half * 8 + i + (i >= 8 ? 8 : 0);
            bv[i] = (f16)(wtrans ? Wb[(long)col * ldw + kw]
                                 : Wb[(long)kw  * ldw + col]);
        }
        acc = wmma_f16(av, bv, acc);
    }
    if (Kmain < K) {
        v16h av, bv;
#pragma unroll
        for (int i = 0; i < 8; ++i) {
            int ka = Kmain + half * 8 + i;      if (ka >= K) ka = K - 1;
            int kb = Kmain + 16 + half * 8 + i; if (kb >= K) kb = K - 1;
            av[i]     = (f16)Arow[ka];
            av[i + 8] = (f16)Arow[kb];
        }
#pragma unroll
        for (int i = 0; i < 16; ++i) {
            int kw = Kmain + half * 8 + i + (i >= 8 ? 8 : 0);
            int kc = (kw >= K) ? K - 1 : kw;
            float w = wtrans ? Wb[(long)col * ldw + kc]
                             : Wb[(long)kc  * ldw + col];
            bv[i] = (f16)((kw < K) ? w : 0.f);        // select, not branch
        }
        acc = wmma_f16(av, bv, acc);
    }

    const int n = tn * 16 + lm;
    if (n >= N) return;
    float* Cfb = Cf + (long)blockIdx.y * sCy + (long)blockIdx.z * sCz;
#pragma unroll
    for (int j = 0; j < 8; ++j) {
        int m = tm * 16 + half * 8 + j;
        if (m >= M) continue;
        Cfb[(long)m * ldc + n] = acc[j] * scale;
    }
}

// ----------------------------- elementwise helpers -------------------------
__global__ void __launch_bounds__(256)
k_posenc_pts(const float* __restrict__ pts, float* __restrict__ out)
{
    int t = blockIdx.x * 256 + threadIdx.x;
    if (t >= NRS_) return;
    float p[3] = { pts[t*3], pts[t*3+1], pts[t*3+2] };
    float* o = out + (long)t * PE_;
    o[0] = p[0]; o[1] = p[1]; o[2] = p[2];
    for (int f = 0; f < NFREQ_; ++f) {
        float fs = exp2f((float)f);
        for (int d = 0; d < 3; ++d) {
            float a = p[d] * fs;
            o[3 + f*6 + d]     = sinf(a);
            o[3 + f*6 + 3 + d] = cosf(a);
        }
    }
}

__global__ void __launch_bounds__(256)
k_posenc_views(const float* __restrict__ rd, float* __restrict__ out)
{
    int r = blockIdx.x * 256 + threadIdx.x;
    if (r >= R_) return;
    float x = rd[r*3], y = rd[r*3+1], z = rd[r*3+2];
    float inv = rsqrtf(x*x + y*y + z*z);
    float p[3] = { x*inv, y*inv, z*inv };
    float* o = out + (long)r * PE_;
    o[0] = p[0]; o[1] = p[1]; o[2] = p[2];
    for (int f = 0; f < NFREQ_; ++f) {
        float fs = exp2f((float)f);
        for (int d = 0; d < 3; ++d) {
            float a = p[d] * fs;
            o[3 + f*6 + d]     = sinf(a);
            o[3 + f*6 + 3 + d] = cosf(a);
        }
    }
}

__global__ void __launch_bounds__(256)
k_maxviews(const float* __restrict__ feat, float* __restrict__ q)
{
    long t = (long)blockIdx.x * 256 + threadIdx.x;
    if (t >= (long)NRS_ * 64) return;
    long rs = t >> 6; int d = (int)(t & 63);
    float m = -1e30f;
    for (int v = 0; v < V_; ++v)
        m = fmaxf(m, feat[(rs * V_ + v) * 64 + d]);
    q[t] = m;
}

// t0 = t0 - qq(broadcast over views) + t1
__global__ void __launch_bounds__(256)
k_kqpf(float* __restrict__ t0, const float* __restrict__ qq, const float* __restrict__ t1)
{
    long t = (long)blockIdx.x * 256 + threadIdx.x;
    if (t >= (long)NRSV_ * 64) return;
    long rsv = t >> 6; int d = (int)(t & 63);
    long rs = rsv / V_;
    t0[t] = t0[t] - qq[rs * 64 + d] + t1[t];
}

// masked softmax over views axis (V_), per (rs,d), in place
__global__ void __launch_bounds__(256)
k_softmax_views(float* __restrict__ a, const unsigned char* __restrict__ mask)
{
    long t = (long)blockIdx.x * 256 + threadIdx.x;
    if (t >= (long)NRS_ * 64) return;
    long rs = t >> 6; int d = (int)(t & 63);
    float vals[V_]; float mx = -1e30f;
    for (int v = 0; v < V_; ++v) {
        float x = mask[rs * V_ + v] ? a[(rs * V_ + v) * 64 + d] : -1e9f;
        vals[v] = x; mx = fmaxf(mx, x);
    }
    float s = 0.f;
    for (int v = 0; v < V_; ++v) { vals[v] = expf(vals[v] - mx); s += vals[v]; }
    float inv = 1.f / s;
    for (int v = 0; v < V_; ++v) a[(rs * V_ + v) * 64 + d] = vals[v] * inv;
}

// xo[rs,d] = sum_v (vv + pf) * a
__global__ void __launch_bounds__(256)
k_reduce_views(const float* __restrict__ vv, const float* __restrict__ pf,
               const float* __restrict__ a, float* __restrict__ xo)
{
    long t = (long)blockIdx.x * 256 + threadIdx.x;
    if (t >= (long)NRS_ * 64) return;
    long rs = t >> 6; int d = (int)(t & 63);
    float s = 0.f;
    for (int v = 0; v < V_; ++v) {
        long i = (rs * V_ + v) * 64 + d;
        s += (vv[i] + pf[i]) * a[i];
    }
    xo[t] = s;
}

// view_sem[rs,c] = sum_v sem[rs,v,c] * a[rs,v,c/8]
__global__ void __launch_bounds__(256)
k_view_sem(const f16* __restrict__ sem, const float* __restrict__ a,
           float* __restrict__ vsem)
{
    long t = (long)blockIdx.x * 256 + threadIdx.x;
    if (t >= (long)NRS_ * SEM_) return;
    long rs = t >> 9; int c = (int)(t & 511);
    float s = 0.f;
    for (int v = 0; v < V_; ++v)
        s += (float)sem[(rs * V_ + v) * SEM_ + c] * a[(rs * V_ + v) * 64 + (c >> 3)];
    vsem[t] = s;
}

// layernorm over last dim 64, one wave per row
__global__ void __launch_bounds__(256)
k_layernorm(const float* __restrict__ X, const float* __restrict__ g,
            const float* __restrict__ b, float* __restrict__ Y, int nrows)
{
    int row = blockIdx.x * 8 + (threadIdx.x >> 5);
    int lane = threadIdx.x & 31;
    if (row >= nrows) return;
    const float* x = X + (long)row * 64;
    float v0 = x[lane], v1 = x[lane + 32];
    float s = v0 + v1, s2 = v0 * v0 + v1 * v1;
    for (int m = 16; m; m >>= 1) { s += __shfl_xor(s, m); s2 += __shfl_xor(s2, m); }
    float mean = s * (1.f / 64.f);
    float var  = s2 * (1.f / 64.f) - mean * mean;
    float r = rsqrtf(var + 1e-6f);
    float* y = Y + (long)row * 64;
    y[lane]      = (v0 - mean) * r * g[lane]      + b[lane];
    y[lane + 32] = (v1 - mean) * r * g[lane + 32] + b[lane + 32];
}

// softmax over last dim 64, one wave per row, in place
__global__ void __launch_bounds__(256)
k_softmax_rows(float* __restrict__ X, int nrows)
{
    int row = blockIdx.x * 8 + (threadIdx.x >> 5);
    int lane = threadIdx.x & 31;
    if (row >= nrows) return;
    float* x = X + (long)row * 64;
    float v0 = x[lane], v1 = x[lane + 32];
    float mx = fmaxf(v0, v1);
    for (int m = 16; m; m >>= 1) mx = fmaxf(mx, __shfl_xor(mx, m));
    float e0 = expf(v0 - mx), e1 = expf(v1 - mx);
    float sum = e0 + e1;
    for (int m = 16; m; m >>= 1) sum += __shfl_xor(sum, m);
    float inv = 1.f / sum;
    x[lane] = e0 * inv; x[lane + 32] = e1 * inv;
}

// amean[r,k] = mean_h attn[r,h,0,k]
__global__ void __launch_bounds__(256)
k_attn_mean(const float* __restrict__ attn, float* __restrict__ amean)
{
    int t = blockIdx.x * 256 + threadIdx.x;
    if (t >= NRS_) return;
    int r = t >> 6, k = t & 63;
    float s = 0.f;
    for (int h = 0; h < NH_; ++h)
        s += attn[(long)r * NH_ * 4096 + (long)h * 4096 + k];
    amean[t] = s * (1.f / NH_);
}

// cat = [q | input_pts | input_views]
__global__ void __launch_bounds__(256)
k_concat(const float* __restrict__ q, const float* __restrict__ inpts,
         const float* __restrict__ invw, float* __restrict__ cat)
{
    long t = (long)blockIdx.x * 256 + threadIdx.x;
    if (t >= (long)NRS_ * 190) return;
    long rs = t / 190; int j = (int)(t % 190);
    float v;
    if (j < 64)            v = q[rs * 64 + j];
    else if (j < 64 + 63)  v = inpts[rs * 63 + (j - 64)];
    else                   v = invw[(rs / S_) * 63 + (j - 127)];
    cat[t] = v;
}

// hmean[r,d] = mean_s h[r,s,d]
__global__ void __launch_bounds__(256)
k_hmean(const float* __restrict__ h, float* __restrict__ hmean)
{
    int t = blockIdx.x * 256 + threadIdx.x;
    if (t >= R_ * 64) return;
    int r = t >> 6, d = t & 63;
    float s = 0.f;
    for (int smp = 0; smp < S_; ++smp)
        s += h[((long)r * S_ + smp) * 64 + d];
    hmean[t] = s * (1.f / S_);
}

// out[r,c] = sum_s amean[r,s] * raysem[r,s,c]
__global__ void __launch_bounds__(256)
k_semout(const float* __restrict__ amean, const float* __restrict__ raysem,
         float* __restrict__ out)
{
    long t = (long)blockIdx.x * 256 + threadIdx.x;
    if (t >= (long)R_ * SEM_) return;
    int r = (int)(t >> 9), c = (int)(t & 511);
    float s = 0.f;
    for (int smp = 0; smp < S_; ++smp)
        s += amean[r * S_ + smp] * raysem[((long)r * S_ + smp) * SEM_ + c];
    out[t] = s;
}

// ----------------------------- host-side helpers ---------------------------
struct PW { const f16* p; int nslab; };

struct CTP {
    const float *a1W,*a1b,*a2W,*a2b,*anG,*anB,*ff1W,*ff1b,*ff2W,*ff2b,*fnG,*fnB,
                *kW,*outW,*outb,*pos1W,*pos1b,*pos2W,*pos2b,*qW,*vW;
};
struct STP {
    const float *anG,*anB,*ff1W,*ff1b,*ff2W,*ff2b,*fnG,*fnB,*kW,*outW,*outb,*qW,*vW;
};
struct CTPk { PW q,k,v,pos1,pos2,a1,a2,out,ff1,ff2; };
struct STPk { PW q,k,v,out,ff1,ff2; };

static void gemmP(hipStream_t s, const float* A, long lda, PW w,
                  const float* bias, const float* Res, long ldr,
                  float* Cf, f16* Ch, long ldc, int M, int N, int K, int relu)
{
    if ((N & 63) == 0) {            // 16x64 strip per wave
        long tg = (long)((M + 15) / 16) * (N >> 6);
        dim3 grid((unsigned)((tg + 7) / 8));
        k_gemm_pw4<<<grid, 256, 0, s>>>(A, lda, w.p, w.nslab, bias, Res, ldr,
                                        Cf, Ch, ldc, M, N, K, relu);
    } else {                        // 16x16 tile per wave
        long tiles = (long)((M + 15) / 16) * ((N + 15) / 16);
        dim3 grid((unsigned)((tiles + 7) / 8));
        k_gemm_pw<<<grid, 256, 0, s>>>(A, lda, w.p, w.nslab, bias, Res, ldr,
                                       Cf, Ch, ldc, M, N, K, relu);
    }
}
// common case: contiguous A (lda=K), f32 out (ldc=N), residual ldr=N
static void gemmW(hipStream_t s, const float* A, PW w, const float* bias,
                  const float* Res, float* C, int M, int N, int K, int relu)
{
    gemmP(s, A, K, w, bias, Res, N, C, nullptr, N, M, N, K, relu);
}

// ---------------------------------------------------------------------------
extern "C" void kernel_launch(void* const* d_in, const int* in_sizes, int n_in,
                              void* d_out, int out_size, void* d_ws, size_t ws_size,
                              hipStream_t stream)
{
    (void)in_sizes; (void)n_in; (void)out_size; (void)ws_size;

    // ---- inputs (setup_inputs dict order) ----
    const float* rgb_feat = (const float*)d_in[0];           // [R,S,V,35]
    const float* dsf      = (const float*)d_in[1];           // [R,S,V,512]
    const float* ray_diff = (const float*)d_in[2];           // [R,S,V,4]
    const unsigned char* mask = (const unsigned char*)d_in[3];
    const float* pts      = (const float*)d_in[4];           // [R,S,3]
    const float* ray_d    = (const float*)d_in[5];           // [R,3]

    // ---- params: JAX pytree leaves, dict keys sorted alphabetically ----
    int p = 6;
    auto nxt = [&]() { return (const float*)d_in[p++]; };
    const float* adW = nxt(); const float* adb = nxt();
    CTP ct[DEPTH_]; STP st[DEPTH_];
    const float *qfc0W[DEPTH_]={0}, *qfc0b[DEPTH_]={0}, *qfc1W[DEPTH_]={0}, *qfc1b[DEPTH_]={0};
    for (int i = 0; i < DEPTH_; ++i) {
        CTP& c = ct[i];
        c.a1W=nxt(); c.a1b=nxt(); c.a2W=nxt(); c.a2b=nxt();
        c.anG=nxt(); c.anB=nxt();
        c.ff1W=nxt(); c.ff1b=nxt(); c.ff2W=nxt(); c.ff2b=nxt();
        c.fnG=nxt(); c.fnB=nxt();
        c.kW=nxt();
        c.outW=nxt(); c.outb=nxt();
        c.pos1W=nxt(); c.pos1b=nxt(); c.pos2W=nxt(); c.pos2b=nxt();
        c.qW=nxt(); c.vW=nxt();
        if (i % 2 == 0) { qfc0W[i]=nxt(); qfc0b[i]=nxt(); qfc1W[i]=nxt(); qfc1b[i]=nxt(); }
        STP& t = st[i];
        t.anG=nxt(); t.anB=nxt(); t.ff1W=nxt(); t.ff1b=nxt(); t.ff2W=nxt(); t.ff2b=nxt();
        t.fnG=nxt(); t.fnB=nxt(); t.kW=nxt(); t.outW=nxt(); t.outb=nxt(); t.qW=nxt(); t.vW=nxt();
    }
    const float* normG = nxt(); const float* normB = nxt();
    const float* rgbW  = nxt(); const float* rgbb  = nxt();
    const float* rf1W  = nxt(); const float* rf1b  = nxt();
    const float* rf2W  = nxt(); const float* rf2b  = nxt();

    // ---- workspace layout ----
    char* base = (char*)d_ws;
    size_t off = 0;
    auto alloc = [&](size_t bytes) -> void* {
        void* ptr = base + off;
        off += (bytes + 255) & ~(size_t)255;
        return ptr;
    };
    float* feat   = (float*)alloc(sizeof(float) * (size_t)NRSV_ * 64);
    f16*   sem    = (f16*)  alloc(sizeof(f16)   * (size_t)NRSV_ * SEM_);
    float* qb     = (float*)alloc(sizeof(float) * (size_t)NRS_ * 64);
    float* inpts  = (float*)alloc(sizeof(float) * (size_t)NRS_ * PE_);
    float* invw   = (float*)alloc(sizeof(float) * (size_t)R_ * PE_);
    float* amean  = (float*)alloc(sizeof(float) * (size_t)NRS_);
    float* raysem = (float*)alloc(sizeof(float) * (size_t)NRS_ * SEM_);
    float* t0     = (float*)alloc(sizeof(float) * (size_t)NRSV_ * 64);
    float* t1     = (float*)alloc(sizeof(float) * (size_t)NRSV_ * 64);
    float* t2     = (float*)alloc(sizeof(float) * (size_t)NRSV_ * 64);
    float* t3     = (float*)alloc(sizeof(float) * (size_t)NRSV_ * 8);
    float* vsem   = (float*)alloc(sizeof(float) * (size_t)NRS_ * SEM_);
    float* xo     = (float*)alloc(sizeof(float) * (size_t)NRS_ * 64);
    float* xb     = (float*)alloc(sizeof(float) * (size_t)NRS_ * 64);
    float* ffh    = (float*)alloc(sizeof(float) * (size_t)NRS_ * FF_);
    float* qh     = (float*)alloc(sizeof(float) * (size_t)NRS_ * 64);
    float* kh     = (float*)alloc(sizeof(float) * (size_t)NRS_ * 64);
    float* vh     = (float*)alloc(sizeof(float) * (size_t)NRS_ * 64);
    float* oh     = (float*)alloc(sizeof(float) * (size_t)NRS_ * 64);
    float* attn   = (float*)alloc(sizeof(float) * (size_t)R_ * NH_ * S_ * S_);
    float* cat    = (float*)alloc(sizeof(float) * (size_t)NRS_ * 190);
    float* hmean  = (float*)alloc(sizeof(float) * (size_t)R_ * 64);

    auto blocks = [](long n) { return (unsigned)((n + 255) / 256); };

    // ---- pack all weights into WMMA B-fragment layout (f16) ----
    auto packW = [&](const float* W, int K, int N) -> PW {
        int tN = (N + 15) / 16, nslab = (K + 31) / 32;
        long nentries = (long)tN * nslab * 32;
        f16* buf = (f16*)alloc((size_t)nentries * 16 * sizeof(f16));
        k_packW<<<blocks(nentries), 256, 0, stream>>>(W, buf, K, N, nslab, nentries);
        return PW{buf, nslab};
    };
    PW Prf1 = packW(rf1W, 35, 64);
    PW Prf2 = packW(rf2W, 64, 64);
    PW Pad  = packW(adW, SEM_, SEM_);
    PW Prgb = packW(rgbW, 64, 3);
    CTPk pct[DEPTH_]; STPk pst[DEPTH_]; PW pqfc0[DEPTH_], pqfc1[DEPTH_];
    for (int i = 0; i < DEPTH_; ++i) {
        const CTP& c = ct[i];
        pct[i].q    = packW(c.qW,   64, 64);
        pct[i].k    = packW(c.kW,   64, 64);
        pct[i].v    = packW(c.vW,   64, 64);
        pct[i].pos1 = packW(c.pos1W, 4, 8);
        pct[i].pos2 = packW(c.pos2W, 8, 64);
        pct[i].a1   = packW(c.a1W,  64, 8);
        pct[i].a2   = packW(c.a2W,   8, 64);
        pct[i].out  = packW(c.outW, 64, 64);
        pct[i].ff1  = packW(c.ff1W, 64, FF_);
        pct[i].ff2  = packW(c.ff2W, FF_, 64);
        if (i % 2 == 0) {
            pqfc0[i] = packW(qfc0W[i], 190, 64);
            pqfc1[i] = packW(qfc1W[i], 64, 64);
        }
        const STP& t = st[i];
        pst[i].q   = packW(t.qW,   64, 64);
        pst[i].k   = packW(t.kW,   64, 64);
        pst[i].v   = packW(t.vW,   64, 64);
        pst[i].out = packW(t.outW, 64, 64);
        pst[i].ff1 = packW(t.ff1W, 64, FF_);
        pst[i].ff2 = packW(t.ff2W, FF_, 64);
    }

    // ---- prep ----
    k_posenc_pts <<<blocks(NRS_), 256, 0, stream>>>(pts, inpts);
    k_posenc_views<<<blocks(R_),  256, 0, stream>>>(ray_d, invw);
    // feat = mlp2(rgb_feat, rgbfeat1, rgbfeat2)
    gemmW(stream, rgb_feat, Prf1, rf1b, nullptr, t0, NRSV_, 64, 35, 1);
    gemmW(stream, t0, Prf2, rf2b, nullptr, feat, NRSV_, 64, 64, 0);
    // sem = relu(dsf @ adaptor)  -> stored f16 (read 8x, halves HBM traffic)
    gemmP(stream, dsf, SEM_, Pad, adb, nullptr, 0, nullptr, sem, SEM_,
          NRSV_, SEM_, SEM_, 1);
    // q = feat.max(views)
    k_maxviews<<<blocks((long)NRS_ * 64), 256, 0, stream>>>(feat, qb);

    for (int L = 0; L < DEPTH_; ++L) {
        const CTP& c = ct[L]; const CTPk& pc = pct[L];
        // ---------------- cross_trans ----------------
        k_layernorm<<<blocks(NRS_ * 8), 256, 0, stream>>>(qb, c.anG, c.anB, xb, NRS_);
        gemmW(stream, xb, pc.q, nullptr, nullptr, xo, NRS_, 64, 64, 0);        // qq -> xo
        gemmW(stream, feat, pc.k, nullptr, nullptr, t0, NRSV_, 64, 64, 0);     // kk
        gemmW(stream, ray_diff, pc.pos1, c.pos1b, nullptr, t3, NRSV_, 8, 4, 1);
        gemmW(stream, t3, pc.pos2, c.pos2b, nullptr, t1, NRSV_, 64, 8, 0);     // pf
        k_kqpf<<<blocks((long)NRSV_ * 64), 256, 0, stream>>>(t0, xo, t1);      // kk-qq+pf
        gemmW(stream, t0, pc.a1, c.a1b, nullptr, t3, NRSV_, 8, 64, 1);
        gemmW(stream, t3, pc.a2, c.a2b, nullptr, t2, NRSV_, 64, 8, 0);         // a logits
        k_softmax_views<<<blocks((long)NRS_ * 64), 256, 0, stream>>>(t2, mask);
        gemmW(stream, feat, pc.v, nullptr, nullptr, t0, NRSV_, 64, 64, 0);     // vv
        k_reduce_views<<<blocks((long)NRS_ * 64), 256, 0, stream>>>(t0, t1, t2, xo);
        k_view_sem<<<blocks((long)NRS_ * SEM_), 256, 0, stream>>>(sem, t2, vsem);
        gemmW(stream, xo, pc.out, c.outb, qb, xb, NRS_, 64, 64, 0);            // xo2 -> xb
        k_layernorm<<<blocks(NRS_ * 8), 256, 0, stream>>>(xb, c.fnG, c.fnB, xo, NRS_);
        gemmW(stream, xo, pc.ff1, c.ff1b, nullptr, ffh, NRS_, FF_, 64, 1);
        gemmW(stream, ffh, pc.ff2, c.ff2b, xb, qb, NRS_, 64, FF_, 0);          // q

        // ---------------- qfc on even layers ----------------
        if (L % 2 == 0) {
            k_concat<<<blocks((long)NRS_ * 190), 256, 0, stream>>>(qb, inpts, invw, cat);
            gemmW(stream, cat, pqfc0[L], qfc0b[L], nullptr, xb, NRS_, 64, 190, 1);
            gemmW(stream, xb, pqfc1[L], qfc1b[L], nullptr, qb, NRS_, 64, 64, 0);
        }

        // ---------------- self_trans ----------------
        const STP& t = st[L]; const STPk& ps = pst[L];
        k_layernorm<<<blocks(NRS_ * 8), 256, 0, stream>>>(qb, t.anG, t.anB, xb, NRS_);
        gemmW(stream, xb, ps.q, nullptr, nullptr, qh, NRS_, 64, 64, 0);
        gemmW(stream, xb, ps.k, nullptr, nullptr, kh, NRS_, 64, 64, 0);
        gemmW(stream, xb, ps.v, nullptr, nullptr, vh, NRS_, 64, 64, 0);
        // logits[r,h] = (Q @ K^T) / 4 : batched over (y=r, z=h), W transposed
        {
            dim3 g(2, R_, NH_);   // 16 tiles / 8 waves
            k_gemm<<<g, 256, 0, stream>>>(
                qh, 64, (long)S_ * 64, 16,
                kh, 64, (long)S_ * 64, 16, 1,
                attn, 64, (long)NH_ * S_ * S_, (long)S_ * S_,
                64, 64, 16, 0.25f);
        }
        k_softmax_rows<<<blocks((long)R_ * NH_ * S_ * 8), 256, 0, stream>>>(attn, R_ * NH_ * S_);
        k_attn_mean<<<blocks(NRS_), 256, 0, stream>>>(attn, amean);
        // out_heads = attn @ V  (merged head layout)
        {
            dim3 g(1, R_, NH_);   // 4 tiles
            k_gemm<<<g, 256, 0, stream>>>(
                attn, 64, (long)NH_ * S_ * S_, (long)S_ * S_,
                vh, 64, (long)S_ * 64, 16, 0,
                oh, 64, (long)S_ * 64, 16,
                64, 16, 64, 1.f);
        }
        // ray_sem = attn @ hd(view_sem) (merged)
        {
            dim3 g(4, R_, NH_);   // 32 tiles
            k_gemm<<<g, 256, 0, stream>>>(
                attn, 64, (long)NH_ * S_ * S_, (long)S_ * S_,
                vsem, SEM_, (long)S_ * SEM_, 128, 0,
                raysem, SEM_, (long)S_ * SEM_, 128,
                64, 128, 64, 1.f);
        }
        gemmW(stream, oh, ps.out, t.outb, qb, xb, NRS_, 64, 64, 0);            // out -> xb
        k_layernorm<<<blocks(NRS_ * 8), 256, 0, stream>>>(xb, t.fnG, t.fnB, xo, NRS_);
        gemmW(stream, xo, ps.ff1, t.ff1b, nullptr, ffh, NRS_, FF_, 64, 1);
        gemmW(stream, ffh, ps.ff2, t.ff2b, xb, qb, NRS_, 64, FF_, 0);          // q
    }

    // ---- head ----
    k_layernorm<<<blocks(NRS_ * 8), 256, 0, stream>>>(qb, normG, normB, xb, NRS_);
    k_hmean<<<blocks(R_ * 64), 256, 0, stream>>>(xb, hmean);
    gemmW(stream, hmean, Prgb, rgbb, nullptr, (float*)d_out, R_, 3, 64, 0);    // rgb
    k_semout<<<blocks((long)R_ * SEM_), 256, 0, stream>>>(amean, raysem,
                                                          (float*)d_out + R_ * 3);
}
